// PCENetwork_377957122303
// MI455X (gfx1250) — compile-verified, matrix operand order checked by reading, not run on hardware
//
#include <hip/hip_runtime.h>
#include <hip/hip_bf16.h>
#include <math.h>

// ---------------------------------------------------------------------------
// PCE network on gfx1250: patch-local 3x3 expert convs as im2col WMMA GEMMs.
// Weights staged into LDS per block via the Tensor Data Mover (TDM).
// Data layout between layers: f16, [b][p][c][pixel(0..63)]  (p = patch 0..255)
// ---------------------------------------------------------------------------

typedef __attribute__((ext_vector_type(16))) _Float16     v16h;
typedef __attribute__((ext_vector_type(8)))  float        v8f;
typedef __attribute__((ext_vector_type(4)))  unsigned int v4u;
typedef __attribute__((ext_vector_type(4)))  int          v4i;
typedef __attribute__((ext_vector_type(8)))  int          v8i;

#define NPATCH 256          // patches per image (16x16 grid of 8x8)
#define BATCH  16
#define QTOT   (BATCH * NPATCH)   // 4096 global patches
#define NEXP   8

// ------------------------- input f32 -> patch f16 --------------------------
__global__ void cvt_input_kernel(const float* __restrict__ X,
                                 _Float16* __restrict__ Xp) {
    // X: [16][3][128][128] f32 -> Xp: [b][p][c][64] f16
    const int total = BATCH * 3 * 128 * 128;
    for (int i = blockIdx.x * blockDim.x + threadIdx.x; i < total;
         i += gridDim.x * blockDim.x) {
        int b  = i / (3 * 16384);
        int r  = i % (3 * 16384);
        int c  = r / 16384;
        int yx = r % 16384;
        int y = yx >> 7, x = yx & 127;
        int p = ((y >> 3) << 4) + (x >> 3);
        int pix = ((y & 7) << 3) + (x & 7);
        Xp[(((size_t)(b * NPATCH + p)) * 3 + c) * 64 + pix] = (_Float16)X[i];
    }
}

// --------------------- weights f32 OIHW -> Bt f16 [N][Kp] ------------------
__global__ void prep_weights_kernel(const float* __restrict__ ew,
                                    _Float16* __restrict__ Bt,
                                    int N, int K, int KP) {
    int total = N * KP;
    for (int i = blockIdx.x * blockDim.x + threadIdx.x; i < total;
         i += gridDim.x * blockDim.x) {
        int n = i / KP, k = i % KP;
        // ew is [N][Cin][3][3]; k = ci*9 + ky*3 + kx is exactly the inner layout
        Bt[i] = (k < K) ? (_Float16)ew[n * K + k] : (_Float16)0.0f;
    }
}

// --------------------------- router (per layer) -----------------------------
// scores[p][e] for p in 0..255 — depends only on coordinates, not on X/batch.
__global__ void router_kernel(const float* __restrict__ rw,   // [16][52]
                              const float* __restrict__ rb,   // [16]
                              const float* __restrict__ rlw,  // [8][16]
                              const float* __restrict__ rlb,  // [8]
                              float* __restrict__ scores) {   // [256][8]
    int p = threadIdx.x;          // one thread per patch, 256 threads
    int ph = p >> 4, pw = p & 15;
    float z[16];
#pragma unroll
    for (int r = 0; r < 16; ++r) z[r] = 0.0f;

    for (int py = 0; py < 8; ++py) {
        for (int px = 0; px < 8; ++px) {
            int y = ph * 8 + py, x = pw * 8 + px;
            float base[4] = { (x + 0.5f) / 128.0f, (y + 0.5f) / 128.0f,
                              (px + 0.5f) / 8.0f,  (py + 0.5f) / 8.0f };
            float cf[52];
#pragma unroll
            for (int j = 0; j < 4; ++j) cf[j] = base[j];
#pragma unroll
            for (int f = 0; f < 6; ++f) {
                float w = 3.14159265358979f * (float)(1 << f);
#pragma unroll
                for (int j = 0; j < 4; ++j) {
                    cf[4 + 8 * f + j]     = sinf(w * base[j]);
                    cf[4 + 8 * f + 4 + j] = cosf(w * base[j]);
                }
            }
#pragma unroll
            for (int r = 0; r < 16; ++r) {
                float s = rb[r];
                const float* wr = rw + r * 52;
                for (int c = 0; c < 52; ++c) s += wr[c] * cf[c];
                z[r] += fmaxf(s, 0.0f);
            }
        }
    }
    float logits[NEXP];
#pragma unroll
    for (int e = 0; e < NEXP; ++e) {
        float s = rlb[e];
#pragma unroll
        for (int r = 0; r < 16; ++r) s += (z[r] * (1.0f / 64.0f)) * rlw[e * 16 + r];
        logits[e] = s;
    }
    float mx = logits[0];
#pragma unroll
    for (int e = 1; e < NEXP; ++e) mx = fmaxf(mx, logits[e]);
    float pe[NEXP], sum = 0.0f;
#pragma unroll
    for (int e = 0; e < NEXP; ++e) { pe[e] = expf(logits[e] - mx); sum += pe[e]; }
    float inv = 1.0f / sum, ksum = 0.0f;
#pragma unroll
    for (int e = 0; e < NEXP; ++e) {
        pe[e] *= inv;
        if (pe[e] < 0.05f) pe[e] = 0.0f;
        ksum += pe[e];
    }
    float rinv = 1.0f / (ksum + 1e-9f);
#pragma unroll
    for (int e = 0; e < NEXP; ++e) scores[p * NEXP + e] = pe[e] * rinv;
}

// ----------------------- expert conv via WMMA GEMM --------------------------
// One block = 2 patches (M = 128 im2col rows), 256 threads = 8 waves.
// LDS: [A im2col tile | full Bt weight tile]. Bt staged by TDM (wave 0),
// overlapped with the cooperative im2col fill; waves then sweep n-tiles with
// chained v_wmma_f32_16x16x32_f16. Epilogue: bias + relu + score-combine.
template <int CIN, int COUT>
__global__ void conv_wmma_kernel(const _Float16* __restrict__ Xin,   // [q][CIN][64]
                                 const _Float16* __restrict__ Bt,    // [N][KP]
                                 const float*    __restrict__ eb,    // [N]
                                 const float*    __restrict__ scores,// [256][8]
                                 _Float16* __restrict__ Xout) {      // [q][COUT][64]
    constexpr int K  = CIN * 9;
    constexpr int KP = ((K + 31) / 32) * 32;
    constexpr int N  = NEXP * COUT;
    constexpr int G  = 2;          // patches per block
    constexpr int M  = G * 64;     // 128 rows
    constexpr int SC = (COUT + 15) / 16;   // output column sets (1 or 2)

    extern __shared__ _Float16 dynlds[];
    _Float16* Alds = dynlds;             // M*KP halves
    _Float16* Blds = dynlds + M * KP;    // N*KP halves

    const int tid = threadIdx.x;
    const int q0  = blockIdx.x * G;        // global patch base (0..4095)

    // ---- wave 0: TDM-load the whole Bt[N][KP] f16 tile into LDS ------------
    if (tid < 32) {
        unsigned long long ga = (unsigned long long)Bt;
        const unsigned ldsB_off = (unsigned)(M * KP * 2);  // bytes from LDS base
        v4u g0;
        g0[0] = 1u;                                   // count=1, user descriptor
        g0[1] = ldsB_off;                             // lds_addr
        g0[2] = (unsigned)(ga & 0xFFFFFFFFu);         // global_addr[31:0]
        g0[3] = (unsigned)((ga >> 32) & 0x01FFFFFFu)  // global_addr[56:32]
                | (2u << 30);                         // type = 2 ("image")
        v8i g1;
        g1[0] = (int)(1u << 16);                      // data_size = 1 (2 bytes)
        g1[1] = (int)(((unsigned)KP & 0xFFFFu) << 16);// tensor_dim0 = KP
        g1[2] = (int)(((unsigned)N  & 0xFFFFu) << 16);// tensor_dim1 = N
        g1[3] = (int)((unsigned)KP << 16);            // tile_dim0 = KP
        g1[4] = (int)N;                               // tile_dim1 = N, tile_dim2=0
        g1[5] = (int)KP;                              // tensor_dim0_stride = KP
        g1[6] = 0;                                    // dim1_stride lo (unused)
        g1[7] = 0;
        v4i g2; g2[0] = 0; g2[1] = 0; g2[2] = 0; g2[3] = 0;
        v4i g3; g3[0] = 0; g3[1] = 0; g3[2] = 0; g3[3] = 0;
#if defined(__clang_major__) && (__clang_major__ >= 23)
        v8i g4; g4[0] = 0; g4[1] = 0; g4[2] = 0; g4[3] = 0;
        g4[4] = 0; g4[5] = 0; g4[6] = 0; g4[7] = 0;
        __builtin_amdgcn_tensor_load_to_lds(g0, g1, g2, g3, g4, 0);
#else
        __builtin_amdgcn_tensor_load_to_lds(g0, g1, g2, g3, 0);
#endif
    }

    // ---- all threads: build im2col A tile (zero-padded borders & K tail) ---
    for (int i = tid; i < M * KP; i += 256) {
        int row = i / KP, k = i % KP;
        int g = row >> 6, pix = row & 63;
        int py = pix >> 3, px = pix & 7;
        float v = 0.0f;
        if (k < K) {
            int ci = k / 9, rem = k % 9;
            int ky = rem / 3, kx = rem % 3;
            int iy = py + ky - 1, ix = px + kx - 1;
            if (iy >= 0 && iy < 8 && ix >= 0 && ix < 8) {
                v = (float)Xin[(((size_t)(q0 + g)) * CIN + ci) * 64 + iy * 8 + ix];
            }
        }
        Alds[i] = (_Float16)v;
    }
    if (tid < 32) __builtin_amdgcn_s_wait_tensorcnt(0);
    __syncthreads();

    const int wave = tid >> 5;
    const int lane = tid & 31;
    const int n16  = lane & 15;        // column within tile / row within tile
    const int hi   = lane >> 4;

    // ISA 7.12.2 16-bit fragment K pattern: dword v -> K pair offset
    int kkofs[8];
#pragma unroll
    for (int v = 0; v < 8; ++v)
        kkofs[v] = ((v & 3) << 1) + (hi << 3) + ((v >> 2) << 4);

    const int mt   = wave;                 // 8 m-tiles, 8 waves
    const int q    = q0 + (mt >> 2);       // patch of this m-tile
    const int pimg = q & (NPATCH - 1);     // patch index within image

    float comb[SC][8];
#pragma unroll
    for (int s = 0; s < SC; ++s)
#pragma unroll
        for (int r = 0; r < 8; ++r) comb[s][r] = 0.0f;

    const _Float16* Arow = Alds + (size_t)(mt * 16 + n16) * KP;

    for (int nt = 0; nt < N / 16; ++nt) {
        const int ncol = nt * 16 + n16;
        const _Float16* Brow = Blds + (size_t)ncol * KP;

        union { v8f v; float f[8]; } acc;
#pragma unroll
        for (int r = 0; r < 8; ++r) acc.f[r] = 0.0f;

        for (int k0 = 0; k0 < KP; k0 += 32) {
            union { v16h h; unsigned u[8]; } af, bf;
#pragma unroll
            for (int v = 0; v < 8; ++v)
                af.u[v] = *(const unsigned*)(Arow + k0 + kkofs[v]);
#pragma unroll
            for (int v = 0; v < 8; ++v)
                bf.u[v] = *(const unsigned*)(Brow + k0 + kkofs[v]);
            acc.v = __builtin_amdgcn_wmma_f32_16x16x32_f16(
                false, af.h, false, bf.h, (short)0, acc.v, false, false);
        }

        const int e   = ncol / COUT;
        const int co  = ncol % COUT;
        const int set = co >> 4;
        const float s    = scores[pimg * NEXP + e];
        const float bias = eb[ncol];
#pragma unroll
        for (int r = 0; r < 8; ++r)
            comb[set][r] += s * fmaxf(acc.f[r] + bias, 0.0f);
    }

    if (COUT == 8) {
        // lanes n and n^8 hold the same output channel, different experts
#pragma unroll
        for (int r = 0; r < 8; ++r)
            comb[0][r] += __shfl_xor(comb[0][r], 8, 32);
    }

    // ---- store combined [16 x COUT] tile, f16 patch layout ----
#pragma unroll
    for (int s = 0; s < SC; ++s) {
        int co = (COUT >= 16) ? (s * 16 + n16) : (n16 & 7);
        bool do_store = (COUT == 8) ? (n16 < 8) : true;
        if (do_store) {
#pragma unroll
            for (int r = 0; r < 8; ++r) {
                int mrow = mt * 16 + r + 8 * hi;
                int pix  = mrow & 63;
                Xout[(((size_t)q) * COUT + co) * 64 + pix] = (_Float16)comb[s][r];
            }
        }
    }
}

// ------------------------- 16x16 mean pool + FC -----------------------------
__global__ void pool_fc_kernel(const _Float16* __restrict__ Xl,  // [q][32][64]
                               const float* __restrict__ fcw,    // [100][2048]
                               const float* __restrict__ fcb,    // [100]
                               float* __restrict__ out) {        // [16][100]
    __shared__ float pooled[2048];
    const int b = blockIdx.x, t = threadIdx.x;
    for (int f = t; f < 2048; f += 256) {
        int c = f >> 6, ij = f & 63, i = ij >> 3, j = ij & 7;
        float s = 0.0f;
#pragma unroll
        for (int di = 0; di < 2; ++di)
#pragma unroll
            for (int dj = 0; dj < 2; ++dj) {
                int p = (2 * i + di) * 16 + (2 * j + dj);
                const _Float16* src = Xl + (((size_t)(b * NPATCH + p)) * 32 + c) * 64;
                for (int px = 0; px < 64; ++px) s += (float)src[px];
            }
        pooled[f] = s * (1.0f / 256.0f);
    }
    __syncthreads();
    for (int cls = t; cls < 100; cls += 256) {
        float s = fcb[cls];
        const float* w = fcw + (size_t)cls * 2048;
        for (int f = 0; f < 2048; ++f) s += pooled[f] * w[f];
        out[b * 100 + cls] = s;
    }
}

// ---------------------------------------------------------------------------
extern "C" void kernel_launch(void* const* d_in, const int* in_sizes, int n_in,
                              void* d_out, int out_size, void* d_ws, size_t ws_size,
                              hipStream_t stream) {
    (void)in_sizes; (void)n_in; (void)out_size; (void)ws_size;

    const float* X = (const float*)d_in[0];
    // per-layer params: ew, eb, rw, rb, rlw, rlb at 1 + 6*li
    const float* ew[4];  const float* eb[4];
    const float* rw[4];  const float* rb[4];
    const float* rlw[4]; const float* rlb[4];
    for (int li = 0; li < 4; ++li) {
        ew[li]  = (const float*)d_in[1 + 6 * li + 0];
        eb[li]  = (const float*)d_in[1 + 6 * li + 1];
        rw[li]  = (const float*)d_in[1 + 6 * li + 2];
        rb[li]  = (const float*)d_in[1 + 6 * li + 3];
        rlw[li] = (const float*)d_in[1 + 6 * li + 4];
        rlb[li] = (const float*)d_in[1 + 6 * li + 5];
    }
    const float* fcw = (const float*)d_in[25];
    const float* fcb = (const float*)d_in[26];
    float* out = (float*)d_out;

    // workspace carve-up
    char* ws = (char*)d_ws;
    _Float16* Xa = (_Float16*)ws;                           // up to 4096*32*64 f16 = 16MB
    _Float16* Xb = (_Float16*)(ws + (size_t)(16 << 20));    // up to 4096*16*64 f16 = 8MB
    _Float16* Bt = (_Float16*)(ws + (size_t)(24 << 20));    // 256*160 f16 = 80KB
    float* scores = (float*)(ws + (size_t)(24 << 20) + (size_t)(128 << 10)); // 8KB

    const int blocks = QTOT / 2;  // 2 patches per block

    // dynamic LDS bytes per layer: (M*KP + N*KP) * sizeof(f16), M = 128
    const size_t lds0 = (size_t)(128 * 32  + 64  * 32)  * 2;  //  12 KB
    const size_t lds1 = (size_t)(128 * 96  + 128 * 96)  * 2;  //  48 KB
    const size_t lds2 = (size_t)(128 * 160 + 128 * 160) * 2;  //  80 KB
    const size_t lds3 = (size_t)(128 * 160 + 256 * 160) * 2;  // 120 KB

    // input -> f16 patch layout
    cvt_input_kernel<<<1024, 256, 0, stream>>>(X, Xa);

    // Layer 0: 3 -> 8   (K=27, KP=32, N=64)
    prep_weights_kernel<<<64, 256, 0, stream>>>(ew[0], Bt, 64, 27, 32);
    router_kernel<<<1, 256, 0, stream>>>(rw[0], rb[0], rlw[0], rlb[0], scores);
    conv_wmma_kernel<3, 8><<<blocks, 256, lds0, stream>>>(Xa, Bt, eb[0], scores, Xb);

    // Layer 1: 8 -> 16  (K=72, KP=96, N=128)
    prep_weights_kernel<<<64, 256, 0, stream>>>(ew[1], Bt, 128, 72, 96);
    router_kernel<<<1, 256, 0, stream>>>(rw[1], rb[1], rlw[1], rlb[1], scores);
    conv_wmma_kernel<8, 16><<<blocks, 256, lds1, stream>>>(Xb, Bt, eb[1], scores, Xa);

    // Layer 2: 16 -> 16 (K=144, KP=160, N=128)
    prep_weights_kernel<<<64, 256, 0, stream>>>(ew[2], Bt, 128, 144, 160);
    router_kernel<<<1, 256, 0, stream>>>(rw[2], rb[2], rlw[2], rlb[2], scores);
    conv_wmma_kernel<16, 16><<<blocks, 256, lds2, stream>>>(Xa, Bt, eb[2], scores, Xb);

    // Layer 3: 16 -> 32 (K=144, KP=160, N=256)
    prep_weights_kernel<<<64, 256, 0, stream>>>(ew[3], Bt, 256, 144, 160);
    router_kernel<<<1, 256, 0, stream>>>(rw[3], rb[3], rlw[3], rlb[3], scores);
    conv_wmma_kernel<16, 32><<<blocks, 256, lds3, stream>>>(Xb, Bt, eb[3], scores, Xa);

    // pool (exact 16x16 mean == 2x2 whole patches) + FC
    pool_fc_kernel<<<BATCH, 256, 0, stream>>>(Xa, fcw, fcb, out);
}
